// ShowAttendTellCore_85882166051169
// MI455X (gfx1250) — compile-verified, matrix operand order checked by reading, not run on hardware
//
#include <hip/hip_runtime.h>
#include <hip/hip_bf16.h>

typedef __attribute__((ext_vector_type(16))) __bf16 v16bf;
typedef __attribute__((ext_vector_type(8)))  __bf16 v8bf;
typedef __attribute__((ext_vector_type(8)))  float  v8f;

#define Bsz 512
#define Lsz 196
#define Fsz 512
#define Asz 512
#define Hsz 512

// ---------------------------------------------------------------------------
// bf16 WMMA fragment helpers (CDNA5 ISA 7.12.2 layouts)
// A 16x32 (MxK): lane = half*16 + m; per-lane bf16 elems are
//   K = half*8 + [0..7]  and  K = 16 + half*8 + [0..7]
// B 32x16 (KxN): symmetric, lane holds column n = lane&15, same K pattern.
// C/D 16x16 f32: elem r of v8f = (row r + 8*half, col lane&15).
// ---------------------------------------------------------------------------
__device__ inline v16bf load_frag(const __bf16* p) {
  v8bf lo = *reinterpret_cast<const v8bf*>(p);
  v8bf hi = *reinterpret_cast<const v8bf*>(p + 16);
  v16bf r;
#pragma unroll
  for (int i = 0; i < 8; ++i) { r[i] = lo[i]; r[i + 8] = hi[i]; }
  return r;
}

__device__ inline v8f wmma_bf16(v16bf a, v16bf b, v8f c) {
  return __builtin_amdgcn_wmma_f32_16x16x32_bf16(
      false, a, false, b, (short)0, c, false, false);
}

// 1x4 N-blocked tile: acc[j] += A(16xK) * W_rows[(j*16)..(j*16+15)] for j=0..3.
// A fragment loaded once per K-step and reused by 4 WMMAs.
__device__ inline void wmma_tile4(const __bf16* __restrict__ ap,   // pre-offset per-lane A ptr
                                  const __bf16* __restrict__ wp,   // pre-offset per-lane B ptr
                                  int ldw, int K, v8f acc[4]) {
  for (int k = 0; k < K; k += 32) {
    v16bf a = load_frag(ap + k);
#pragma unroll
    for (int j = 0; j < 4; ++j) {
      v16bf b = load_frag(wp + (size_t)j * 16 * ldw + k);
      acc[j] = wmma_bf16(a, b, acc[j]);
    }
  }
}

// ---------------------------------------------------------------------------
// Converters
// ---------------------------------------------------------------------------
__global__ void cvt_bf16(const float* __restrict__ src, __bf16* __restrict__ dst, int n) {
  int i = blockIdx.x * blockDim.x + threadIdx.x;
  if (i < n) dst[i] = (__bf16)src[i];
}

__global__ void cvt_bf16_strided(const float* __restrict__ src, __bf16* __restrict__ dst,
                                 int n, int cols, int dstride, int doff) {
  int i = blockIdx.x * blockDim.x + threadIdx.x;
  if (i < n) {
    int r = i / cols, c = i - r * cols;
    dst[r * dstride + doff + c] = (__bf16)src[i];
  }
}

// ---------------------------------------------------------------------------
// hid_att[m,n] = sum_k h[m,k]*W_hid[n,k] + b_hid[n] + b_ctx[n]
// 16x64 per wave; M=N=K=512 -> 256 waves -> 32 blocks.
// ---------------------------------------------------------------------------
__global__ void hid_att_kernel(const __bf16* __restrict__ hbf,
                               const __bf16* __restrict__ Whid,
                               const float* __restrict__ b_hid,
                               const float* __restrict__ b_ctx,
                               float* __restrict__ hid_att) {
  int lane = threadIdx.x & 31;
  int w = blockIdx.x * (blockDim.x >> 5) + (threadIdx.x >> 5);
  int nb = w & 7, mt = w >> 3;          // 8 n-blocks of 64, 32 m-tiles
  int half = lane >> 4, col = lane & 15;
  v8f acc[4] = {};
  const __bf16* ap = hbf + (size_t)(mt * 16 + col) * Hsz + half * 8;
  const __bf16* wp = Whid + (size_t)(nb * 64 + col) * Hsz + half * 8;
  wmma_tile4(ap, wp, Hsz, Hsz, acc);
#pragma unroll
  for (int j = 0; j < 4; ++j) {
    int n = nb * 64 + j * 16 + col;
    float bias = b_hid[n] + b_ctx[n];
#pragma unroll
    for (int r = 0; r < 8; ++r) {
      int m = mt * 16 + r + 8 * half;
      hid_att[m * Asz + n] = acc[j][r] + bias;
    }
  }
}

// ---------------------------------------------------------------------------
// Fused score GEMM. One 16-row M-tile per wave; the wave's A tile (16x512
// bf16 = 16KB) is staged to LDS once with async global->LDS, then reused for
// all 32 N-tiles (4-way N-blocked -> 512 WMMAs / wave).
// ---------------------------------------------------------------------------
__global__ void scores_kernel(const __bf16* __restrict__ attbf,
                              const __bf16* __restrict__ Wctx,
                              const float* __restrict__ hid_att,
                              const float* __restrict__ w_score,
                              const float* __restrict__ b_score,
                              float* __restrict__ scores) {
  extern __shared__ __align__(16) __bf16 smem[];    // 8 waves * 16*512 bf16
  int lane = threadIdx.x & 31;
  int wave = threadIdx.x >> 5;
  int w = blockIdx.x * (blockDim.x >> 5) + wave;
  int m0 = w * 16;
  int half = lane >> 4, col = lane & 15;

  __bf16* atile = smem + (size_t)wave * 16 * Fsz;

  // Async-stage the contiguous 16KB A tile: 32 issues x (32 lanes x 16B).
  {
    const char* gsrc = (const char*)(attbf + (size_t)m0 * Fsz);
    unsigned lbase = (unsigned)(uintptr_t)atile;
    for (int it = 0; it < 32; ++it) {
      unsigned loff = lbase + (unsigned)(it * 512 + lane * 16);
      unsigned long long ga =
          (unsigned long long)(uintptr_t)(gsrc + it * 512 + lane * 16);
      asm volatile("global_load_async_to_lds_b128 %0, %1, off"
                   :: "v"(loff), "v"(ga) : "memory");
    }
    asm volatile("s_wait_asynccnt 0x0" ::: "memory");
  }

  const __bf16* ap = atile + (size_t)col * Fsz + half * 8;  // ds_load_b128 path

  float part[8];
#pragma unroll
  for (int r = 0; r < 8; ++r) part[r] = 0.f;

  for (int nb = 0; nb < Asz / 64; ++nb) {
    v8f acc[4] = {};
    const __bf16* wp = Wctx + (size_t)(nb * 64 + col) * Fsz + half * 8;
    wmma_tile4(ap, wp, Fsz, Fsz, acc);
#pragma unroll
    for (int j = 0; j < 4; ++j) {
      int n = nb * 64 + j * 16 + col;
      float wsn = w_score[n];
#pragma unroll
      for (int r = 0; r < 8; ++r) {
        int m = m0 + r + 8 * half;
        int b = m / Lsz;
        part[r] += tanhf(acc[j][r] + hid_att[b * Asz + n]) * wsn;
      }
    }
  }
  // reduce over the 16 lanes that share a row
#pragma unroll
  for (int r = 0; r < 8; ++r) {
    float p = part[r];
    p += __shfl_xor(p, 1, 32);
    p += __shfl_xor(p, 2, 32);
    p += __shfl_xor(p, 4, 32);
    p += __shfl_xor(p, 8, 32);
    part[r] = p;
  }
  if (col == 0) {
    float bs = b_score[0];
#pragma unroll
    for (int r = 0; r < 8; ++r) scores[m0 + r + 8 * half] = part[r] + bs;
  }
}

// ---------------------------------------------------------------------------
// Row softmax over L=196, one block per batch row.
// ---------------------------------------------------------------------------
__global__ void softmax_kernel(const float* __restrict__ scores, float* __restrict__ probs) {
  __shared__ float red[256];
  int b = blockIdx.x, tid = threadIdx.x;
  float v = (tid < Lsz) ? scores[b * Lsz + tid] : -1e30f;
  red[tid] = v;
  __syncthreads();
  for (int s = 128; s > 0; s >>= 1) {
    if (tid < s) red[tid] = fmaxf(red[tid], red[tid + s]);
    __syncthreads();
  }
  float mx = red[0];
  __syncthreads();
  float e = (tid < Lsz) ? __expf(v - mx) : 0.f;
  red[tid] = e;
  __syncthreads();
  for (int s = 128; s > 0; s >>= 1) {
    if (tid < s) red[tid] += red[tid + s];
    __syncthreads();
  }
  float inv = 1.f / red[0];
  if (tid < Lsz) probs[b * Lsz + tid] = e * inv;
}

// ---------------------------------------------------------------------------
// att_res[b,f] = sum_l probs[b,l] * att[b,l,f]; written bf16 into x[:,512+f].
// ---------------------------------------------------------------------------
__global__ void att_res_kernel(const float* __restrict__ probs,
                               const __bf16* __restrict__ attbf,
                               __bf16* __restrict__ xbf) {
  int b = blockIdx.x, tid = threadIdx.x;
  const __bf16* base = attbf + (size_t)b * Lsz * Fsz;
  float a0 = 0.f, a1 = 0.f;
  for (int l = 0; l < Lsz; ++l) {
    float p = probs[b * Lsz + l];
    a0 += p * (float)base[l * Fsz + tid];
    a1 += p * (float)base[l * Fsz + tid + 256];
  }
  xbf[b * (Hsz + Fsz) + Hsz + tid] = (__bf16)a0;
  xbf[b * (Hsz + Fsz) + Hsz + tid + 256] = (__bf16)a1;
}

// ---------------------------------------------------------------------------
// gates[m,n] = x @ W_ih^T + h @ W_hh^T.  16x64 per wave (4-way N-blocked),
// M=512, N=2048 -> 1024 waves -> 128 blocks.  192 WMMAs per wave.
// ---------------------------------------------------------------------------
__global__ void gates_kernel(const __bf16* __restrict__ xbf,
                             const __bf16* __restrict__ Wih,
                             const __bf16* __restrict__ hbf,
                             const __bf16* __restrict__ Whh,
                             float* __restrict__ gates) {
  int lane = threadIdx.x & 31;
  int w = blockIdx.x * (blockDim.x >> 5) + (threadIdx.x >> 5);
  int nb = w & 31, mt = w >> 5;        // 32 n-blocks of 64, 32 m-tiles
  int half = lane >> 4, col = lane & 15;
  v8f acc[4] = {};
  {
    const __bf16* ap = xbf + (size_t)(mt * 16 + col) * (Hsz + Fsz) + half * 8;
    const __bf16* wp = Wih + (size_t)(nb * 64 + col) * (Hsz + Fsz) + half * 8;
    wmma_tile4(ap, wp, Hsz + Fsz, Hsz + Fsz, acc);
  }
  {
    const __bf16* ap = hbf + (size_t)(mt * 16 + col) * Hsz + half * 8;
    const __bf16* wp = Whh + (size_t)(nb * 64 + col) * Hsz + half * 8;
    wmma_tile4(ap, wp, Hsz, Hsz, acc);
  }
#pragma unroll
  for (int j = 0; j < 4; ++j) {
    int n = nb * 64 + j * 16 + col;
#pragma unroll
    for (int r = 0; r < 8; ++r) {
      int m = mt * 16 + r + 8 * half;
      gates[(size_t)m * 4 * Hsz + n] = acc[j][r];
    }
  }
}

// ---------------------------------------------------------------------------
// LSTM pointwise; out = [h | h | c] flat.
// ---------------------------------------------------------------------------
__device__ inline float sigf(float x) { return 1.f / (1.f + __expf(-x)); }

__global__ void lstm_kernel(const float* __restrict__ gates,
                            const float* __restrict__ c0,
                            float* __restrict__ out) {
  int idx = blockIdx.x * blockDim.x + threadIdx.x;  // 0..B*H-1
  int b = idx >> 9, j = idx & 511;
  const float* g = gates + (size_t)b * 4 * Hsz;
  float i = sigf(g[j]);
  float f = sigf(g[Hsz + j]);
  float gg = tanhf(g[2 * Hsz + j]);
  float o = sigf(g[3 * Hsz + j]);
  float c = f * c0[idx] + i * gg;
  float h = o * tanhf(c);
  out[idx] = h;
  out[Bsz * Hsz + idx] = h;
  out[2 * Bsz * Hsz + idx] = c;
}

// ---------------------------------------------------------------------------
extern "C" void kernel_launch(void* const* d_in, const int* in_sizes, int n_in,
                              void* d_out, int out_size, void* d_ws, size_t ws_size,
                              hipStream_t stream) {
  const float* xt       = (const float*)d_in[0];
  // d_in[1] fc_feats: unused by reference
  const float* att      = (const float*)d_in[2];
  const float* h0       = (const float*)d_in[3];
  const float* c0       = (const float*)d_in[4];
  const float* W_ctx    = (const float*)d_in[5];
  const float* b_ctx    = (const float*)d_in[6];
  const float* W_hid    = (const float*)d_in[7];
  const float* b_hid    = (const float*)d_in[8];
  const float* w_score  = (const float*)d_in[9];
  const float* b_score  = (const float*)d_in[10];
  const float* W_ih     = (const float*)d_in[11];
  const float* W_hh     = (const float*)d_in[12];
  float* out = (float*)d_out;

  char* w = (char*)d_ws;
  size_t o = 0;
  __bf16* att_bf  = (__bf16*)(w + o); o += (size_t)Bsz * Lsz * Fsz * 2;    // 102.8 MB
  __bf16* Wctx_bf = (__bf16*)(w + o); o += (size_t)Asz * Fsz * 2;
  __bf16* Whid_bf = (__bf16*)(w + o); o += (size_t)Asz * Hsz * 2;
  __bf16* h_bf    = (__bf16*)(w + o); o += (size_t)Bsz * Hsz * 2;
  __bf16* Wih_bf  = (__bf16*)(w + o); o += (size_t)4 * Hsz * (Hsz + Fsz) * 2;
  __bf16* Whh_bf  = (__bf16*)(w + o); o += (size_t)4 * Hsz * Hsz * 2;
  __bf16* x_bf    = (__bf16*)(w + o); o += (size_t)Bsz * (Hsz + Fsz) * 2;
  float* hid_att  = (float*)(w + o);  o += (size_t)Bsz * Asz * 4;
  float* scores   = (float*)(w + o);  o += (size_t)Bsz * Lsz * 4;
  float* probs    = (float*)(w + o);  o += (size_t)Bsz * Lsz * 4;
  float* gates    = (float*)(w + o);  o += (size_t)Bsz * 4 * Hsz * 4;
  (void)ws_size; (void)n_in; (void)in_sizes; (void)out_size;

  const int T = 256;
  auto blk = [](size_t n) { return (unsigned)((n + 255) / 256); };

  // 1) conversions
  cvt_bf16<<<blk((size_t)Bsz * Lsz * Fsz), T, 0, stream>>>(att, att_bf, Bsz * Lsz * Fsz);
  cvt_bf16<<<blk((size_t)Asz * Fsz), T, 0, stream>>>(W_ctx, Wctx_bf, Asz * Fsz);
  cvt_bf16<<<blk((size_t)Asz * Hsz), T, 0, stream>>>(W_hid, Whid_bf, Asz * Hsz);
  cvt_bf16<<<blk((size_t)Bsz * Hsz), T, 0, stream>>>(h0, h_bf, Bsz * Hsz);
  cvt_bf16<<<blk((size_t)4 * Hsz * (Hsz + Fsz)), T, 0, stream>>>(W_ih, Wih_bf, 4 * Hsz * (Hsz + Fsz));
  cvt_bf16<<<blk((size_t)4 * Hsz * Hsz), T, 0, stream>>>(W_hh, Whh_bf, 4 * Hsz * Hsz);
  cvt_bf16_strided<<<blk((size_t)Bsz * Hsz), T, 0, stream>>>(xt, x_bf, Bsz * Hsz, Hsz, Hsz + Fsz, 0);

  // 2) hid_att (256 waves -> 32 blocks)
  hid_att_kernel<<<32, T, 0, stream>>>(h_bf, Whid_bf, b_hid, b_ctx, hid_att);

  // 3) fused score GEMM: 6272 M-tiles -> 784 blocks * 8 waves, 128KB dyn-LDS
  scores_kernel<<<784, T, 8 * 16 * Fsz * sizeof(__bf16), stream>>>(
      att_bf, Wctx_bf, hid_att, w_score, b_score, scores);

  // 4) softmax over L
  softmax_kernel<<<Bsz, T, 0, stream>>>(scores, probs);

  // 5) att_res -> bf16 into x[:, 512:]
  att_res_kernel<<<Bsz, T, 0, stream>>>(probs, att_bf, x_bf);

  // 6) gates GEMM (1024 waves -> 128 blocks)
  gates_kernel<<<128, T, 0, stream>>>(x_bf, Wih_bf, h_bf, Whh_bf, gates);

  // 7) LSTM pointwise
  lstm_kernel<<<blk((size_t)Bsz * Hsz), T, 0, stream>>>(gates, c0, out);
}
